// EncoderBlock_48713519071569
// MI455X (gfx1250) — compile-verified, matrix operand order checked by reading, not run on hardware
//
#include <hip/hip_runtime.h>
#include <hip/hip_bf16.h>
#include <math.h>

#define S_LEN  2048
#define BATCH  2
#define EMB    1024
#define HEADS  16
#define DHEAD  64
#define DFF    4096
#define MROWS  (BATCH * S_LEN)

typedef __attribute__((ext_vector_type(16))) __bf16 bf16x16;
typedef __attribute__((ext_vector_type(8)))  __bf16 bf16x8;
typedef __attribute__((ext_vector_type(8)))  float  f32x8;
typedef int i32x4 __attribute__((vector_size(16)));   // matches builtin param type

#if defined(__has_builtin)
#if __has_builtin(__builtin_amdgcn_global_load_async_to_lds_b128) && \
    __has_builtin(__builtin_amdgcn_s_wait_asynccnt)
#define HAVE_ASYNC_LDS 1
#endif
#endif
#ifndef HAVE_ASYNC_LDS
#define HAVE_ASYNC_LDS 0
#endif

union Frag16 { bf16x16 v; bf16x8 h[2]; };

static __device__ inline f32x8 wmma_bf16(const Frag16& a, const Frag16& b, f32x8 c) {
  return __builtin_amdgcn_wmma_f32_16x16x32_bf16(false, a.v, false, b.v, (short)0, c,
                                                 false, false);
}

#if HAVE_ASYNC_LDS
// one 32-byte async copy (two b128 ops) global -> LDS, tracked by ASYNCcnt
static __device__ inline void async_copy32(const __bf16* g, __bf16* l) {
  typedef __attribute__((address_space(1))) i32x4 g_i32x4;
  typedef __attribute__((address_space(3))) i32x4 l_i32x4;
  g_i32x4* gp = (g_i32x4*)(g);
  l_i32x4* lp = (l_i32x4*)(l);
  __builtin_amdgcn_global_load_async_to_lds_b128(gp, lp, 0, 0);
  __builtin_amdgcn_global_load_async_to_lds_b128(gp, lp, 16, 0);
}
#endif

// ---------------------------------------------------------------------------
// fp32 -> bf16 conversion / re-layout kernels
// ---------------------------------------------------------------------------

// src [S,B,E] f32 -> xb [B,S,E] bf16
__global__ void cvt_src_kernel(const float* __restrict__ src, __bf16* __restrict__ xb) {
  int i = blockIdx.x * blockDim.x + threadIdx.x;
  if (i >= MROWS * EMB) return;
  int e = i & (EMB - 1);
  int s = (i >> 10) & (S_LEN - 1);
  int b = i >> 21;
  xb[i] = (__bf16)src[((size_t)s * BATCH + b) * EMB + e];
}

// W [H,E,Dh] f32 -> N-major B operand: out[(h*Dh+d)*EMB + e] = W[h,e,d]
__global__ void cvt_qkvw_kernel(const float* __restrict__ w, __bf16* __restrict__ out) {
  int i = blockIdx.x * blockDim.x + threadIdx.x;
  if (i >= EMB * EMB) return;
  int k = i & (EMB - 1);          // e
  int n = i >> 10;                // h*64 + d
  int h = n >> 6, d = n & 63;
  out[i] = (__bf16)w[((size_t)h * EMB + k) * DHEAD + d];
}

// plain copy f32 -> bf16 (proj_W is already N-major for out = a @ proj_W.T)
__global__ void cvt_copy_kernel(const float* __restrict__ in, __bf16* __restrict__ out, int n) {
  int i = blockIdx.x * blockDim.x + threadIdx.x;
  if (i < n) out[i] = (__bf16)in[i];
}

// in [K][N] f32 -> out [N][K] bf16 (N-major B operand)
__global__ void cvt_T_kernel(const float* __restrict__ in, __bf16* __restrict__ out,
                             int K, int N) {
  int i = blockIdx.x * blockDim.x + threadIdx.x;
  if (i >= K * N) return;
  int n = i / K;
  int k = i - n * K;
  out[i] = (__bf16)in[(size_t)k * N + n];
}

// ---------------------------------------------------------------------------
// Tiled WMMA GEMM: C[M,N] = A[M,K](bf16,row-major) @ Bt[N,K](bf16,N-major)
// 128x128 block, 8 wave32 waves, 64x32 per wave, async double-buffered LDS fill.
// FLAGS: 1=bias  2=relu  4=residual-add(f32)  8=store f32  16=store bf16
//        32=store bf16 transposed per batch (vT layout [B][N][S])
// ---------------------------------------------------------------------------
template <int FLAGS>
__global__ __launch_bounds__(256) void gemm_bf16_kernel(
    const __bf16* __restrict__ A, const __bf16* __restrict__ Bt,
    float* __restrict__ Cf, __bf16* __restrict__ Cb,
    const float* __restrict__ bias, const float* __restrict__ resid,
    int M, int N, int K) {
#if HAVE_ASYNC_LDS
  __shared__ __bf16 As[2][128 * 32];
  __shared__ __bf16 Bs[2][128 * 32];
#else
  __shared__ __bf16 As[1][128 * 32];
  __shared__ __bf16 Bs[1][128 * 32];
#endif

  const int tid = threadIdx.x;
  const int lane = tid & 31;
  const int wave = tid >> 5;
  const int wm = wave & 1;        // 2 wave rows  (64 M each)
  const int wn = wave >> 1;       // 4 wave cols  (32 N each)
  const int bm = blockIdx.y * 128;
  const int bn = blockIdx.x * 128;
  const int lrow = lane & 15;
  const int khalf = (lane >> 4) * 8;   // K offset pattern for wmma frags
  const int ldr = tid >> 1;            // tile row this thread loads
  const int ldc = (tid & 1) * 16;      // 16 bf16 (32B) per thread

  f32x8 acc[4][2];
  const f32x8 z = {0.f, 0.f, 0.f, 0.f, 0.f, 0.f, 0.f, 0.f};
#pragma unroll
  for (int i = 0; i < 4; ++i)
#pragma unroll
    for (int j = 0; j < 2; ++j) acc[i][j] = z;

  const __bf16* gA = A + (size_t)(bm + ldr) * K + ldc;
  const __bf16* gB = Bt + (size_t)(bn + ldr) * K + ldc;
  const int ntiles = K >> 5;

#if HAVE_ASYNC_LDS
  // prologue: fill buffer 0 with tile 0 (4 async b128 ops per thread)
  async_copy32(gA, &As[0][ldr * 32 + ldc]);
  async_copy32(gB, &Bs[0][ldr * 32 + ldc]);
#endif

  for (int t = 0; t < ntiles; ++t) {
    const int kt = t << 5;
#if HAVE_ASYNC_LDS
    const int cur = t & 1;
    if (t + 1 < ntiles) {  // fill the other buffer with the next tile
      async_copy32(gA + kt + 32, &As[cur ^ 1][ldr * 32 + ldc]);
      async_copy32(gB + kt + 32, &Bs[cur ^ 1][ldr * 32 + ldc]);
      __builtin_amdgcn_s_wait_asynccnt(4);  // tile t's copies complete
    } else {
      __builtin_amdgcn_s_wait_asynccnt(0);
    }
    __syncthreads();  // tile t visible to all waves
#else
    const int cur = 0;
    __syncthreads();
    {
      const bf16x8* ga = (const bf16x8*)(gA + kt);
      const bf16x8* gb = (const bf16x8*)(gB + kt);
      bf16x8 a0 = ga[0], a1 = ga[1];
      bf16x8 b0 = gb[0], b1 = gb[1];
      bf16x8* sa = (bf16x8*)&As[0][ldr * 32 + ldc];
      bf16x8* sb = (bf16x8*)&Bs[0][ldr * 32 + ldc];
      sa[0] = a0; sa[1] = a1;
      sb[0] = b0; sb[1] = b1;
      if (kt + 32 < K) {
        __builtin_prefetch(gA + kt + 32, 0, 0);
        __builtin_prefetch(gB + kt + 32, 0, 0);
      }
    }
    __syncthreads();
#endif

    Frag16 af[4], bfg[2];
#pragma unroll
    for (int i = 0; i < 4; ++i) {
      const __bf16* p = &As[cur][(wm * 64 + i * 16 + lrow) * 32 + khalf];
      af[i].h[0] = *(const bf16x8*)p;
      af[i].h[1] = *(const bf16x8*)(p + 16);
    }
#pragma unroll
    for (int j = 0; j < 2; ++j) {
      const __bf16* p = &Bs[cur][(wn * 32 + j * 16 + lrow) * 32 + khalf];
      bfg[j].h[0] = *(const bf16x8*)p;
      bfg[j].h[1] = *(const bf16x8*)(p + 16);
    }
#pragma unroll
    for (int i = 0; i < 4; ++i)
#pragma unroll
      for (int j = 0; j < 2; ++j) acc[i][j] = wmma_bf16(af[i], bfg[j], acc[i][j]);

#if HAVE_ASYNC_LDS
    __syncthreads();  // everyone done reading buffer `cur` before it is refilled
#endif
  }

  // epilogue (C layout: VGPR r -> M = r (lanes 0-15) / r+8 (lanes 16-31); N = lane&15)
  const int rsel = (lane < 16) ? 0 : 8;
#pragma unroll
  for (int i = 0; i < 4; ++i) {
#pragma unroll
    for (int j = 0; j < 2; ++j) {
      const int col = bn + wn * 32 + j * 16 + lrow;
      const float bv = (FLAGS & 1) ? bias[col] : 0.f;
#pragma unroll
      for (int r = 0; r < 8; ++r) {
        const int row = bm + wm * 64 + i * 16 + rsel + r;
        float v = acc[i][j][r] + bv;
        if (FLAGS & 2) v = v > 0.f ? v : 0.f;
        if (FLAGS & 4) v += resid[(size_t)row * N + col];
        if (FLAGS & 8) Cf[(size_t)row * N + col] = v;
        if (FLAGS & 16) Cb[(size_t)row * N + col] = (__bf16)v;
        if (FLAGS & 32)
          Cb[((size_t)(row >> 11) * N + col) * S_LEN + (row & (S_LEN - 1))] = (__bf16)v;
      }
    }
  }
}

// ---------------------------------------------------------------------------
// Flash attention: q,k [B,S,H*Dh] bf16 ; vT [B,H*Dh,S] bf16 -> attn [B,S,H*Dh] bf16
// block = 128 threads (4 waves); one block per (b, h, 64 q rows); wave owns 16 rows
// ---------------------------------------------------------------------------
__global__ __launch_bounds__(128) void flash_attn_kernel(
    const __bf16* __restrict__ q, const __bf16* __restrict__ k,
    const __bf16* __restrict__ vT, const float* __restrict__ src_mask,
    const unsigned char* __restrict__ pad, __bf16* __restrict__ attn) {
  __shared__ __bf16 P[4][16 * 32];

  const int tid = threadIdx.x;
  const int lane = tid & 31;
  const int wave = tid >> 5;
  const int qt = blockIdx.x & 31;          // 32 q tiles of 64
  const int bh = blockIdx.x >> 5;
  const int h = bh & (HEADS - 1);
  const int b = bh >> 4;
  const int lrow = lane & 15;
  const int khalf = (lane >> 4) * 8;
  const int rsel = (lane < 16) ? 0 : 8;
  const int qbase = qt * 64 + wave * 16;

  // Q fragments for this wave's 16 rows, all Dh=64 (two K=32 chunks)
  Frag16 qf[2];
  {
    const __bf16* qp = q + ((size_t)(b * S_LEN + qbase + lrow)) * EMB + h * DHEAD;
#pragma unroll
    for (int f = 0; f < 2; ++f) {
      qf[f].h[0] = *(const bf16x8*)(qp + f * 32 + khalf);
      qf[f].h[1] = *(const bf16x8*)(qp + f * 32 + khalf + 16);
    }
  }

  float mrun[8], lrun[8];
  f32x8 accO[4];
  const f32x8 z = {0.f, 0.f, 0.f, 0.f, 0.f, 0.f, 0.f, 0.f};
#pragma unroll
  for (int r = 0; r < 8; ++r) { mrun[r] = -INFINITY; lrun[r] = 0.f; }
#pragma unroll
  for (int j = 0; j < 4; ++j) accO[j] = z;

  const __bf16* vb = vT + (size_t)b * EMB * S_LEN;
  __bf16* pw = &P[wave][0];

  for (int kv = 0; kv < S_LEN; kv += 32) {
    // ---- scores S = (Q K^T) * 1/sqrt(Dh), masked -----------------------
    f32x8 sfr[2];
#pragma unroll
    for (int j = 0; j < 2; ++j) {
      const __bf16* kp = k + ((size_t)(b * S_LEN + kv + j * 16 + lrow)) * EMB + h * DHEAD;
      Frag16 kf0, kf1;
      kf0.h[0] = *(const bf16x8*)(kp + khalf);
      kf0.h[1] = *(const bf16x8*)(kp + khalf + 16);
      kf1.h[0] = *(const bf16x8*)(kp + 32 + khalf);
      kf1.h[1] = *(const bf16x8*)(kp + 32 + khalf + 16);
      f32x8 s = z;
      s = wmma_bf16(qf[0], kf0, s);
      s = wmma_bf16(qf[1], kf1, s);
      const int colg = kv + j * 16 + lrow;
      const bool pd = pad[b * S_LEN + colg] != 0;
#pragma unroll
      for (int r = 0; r < 8; ++r) {
        float sv = s[r] * 0.125f;  // 1/sqrt(64)
        const int rowg = qbase + rsel + r;
        const float mk = src_mask[(size_t)rowg * S_LEN + colg];
        if (pd || mk == -INFINITY) sv = -INFINITY;
        s[r] = sv;
      }
      sfr[j] = s;
    }

    // ---- online softmax (row reductions across lanes 0-15 of each half) -
#pragma unroll
    for (int r = 0; r < 8; ++r) {
      float rm = fmaxf(sfr[0][r], sfr[1][r]);
      rm = fmaxf(rm, __shfl_xor(rm, 1));
      rm = fmaxf(rm, __shfl_xor(rm, 2));
      rm = fmaxf(rm, __shfl_xor(rm, 4));
      rm = fmaxf(rm, __shfl_xor(rm, 8));
      const float mnew = fmaxf(mrun[r], rm);
      const float alpha = __expf(mrun[r] - mnew);
      float p0 = __expf(sfr[0][r] - mnew);
      float p1 = __expf(sfr[1][r] - mnew);
      sfr[0][r] = p0;
      sfr[1][r] = p1;
      float rs = p0 + p1;
      rs += __shfl_xor(rs, 1);
      rs += __shfl_xor(rs, 2);
      rs += __shfl_xor(rs, 4);
      rs += __shfl_xor(rs, 8);
      lrun[r] = lrun[r] * alpha + rs;
      mrun[r] = mnew;
#pragma unroll
      for (int j = 0; j < 4; ++j) accO[j][r] *= alpha;
    }

    // ---- C-layout P -> LDS -> A-layout fragment -------------------------
#pragma unroll
    for (int j = 0; j < 2; ++j)
#pragma unroll
      for (int r = 0; r < 8; ++r)
        pw[(rsel + r) * 32 + j * 16 + lrow] = (__bf16)sfr[j][r];
    Frag16 pa;
    pa.h[0] = *(const bf16x8*)(pw + lrow * 32 + khalf);
    pa.h[1] = *(const bf16x8*)(pw + lrow * 32 + khalf + 16);

    // ---- O += P @ V  (V from transposed layout: B-frag rows = d) --------
#pragma unroll
    for (int d0 = 0; d0 < 4; ++d0) {
      const __bf16* vp = vb + (size_t)(h * DHEAD + d0 * 16 + lrow) * S_LEN + kv;
      Frag16 vf;
      vf.h[0] = *(const bf16x8*)(vp + khalf);
      vf.h[1] = *(const bf16x8*)(vp + khalf + 16);
      accO[d0] = wmma_bf16(pa, vf, accO[d0]);
    }
  }

  // ---- normalize and store (heads already concatenated by layout) -------
#pragma unroll
  for (int d0 = 0; d0 < 4; ++d0)
#pragma unroll
    for (int r = 0; r < 8; ++r) {
      const int rowg = qbase + rsel + r;
      const float o = accO[d0][r] / lrun[r];
      attn[((size_t)(b * S_LEN + rowg)) * EMB + h * DHEAD + d0 * 16 + lrow] = (__bf16)o;
    }
}

// ---------------------------------------------------------------------------
// Residual + LayerNorm kernels (one block = one row of E=1024)
// ---------------------------------------------------------------------------
__global__ __launch_bounds__(256) void add_ln1_kernel(
    const float* __restrict__ src, const float* __restrict__ proj,
    const float* __restrict__ g, const float* __restrict__ bb,
    float* __restrict__ xf, __bf16* __restrict__ xb) {
  __shared__ float red[256];
  const int row = blockIdx.x;       // b*S + s
  const int b = row >> 11;
  const int s = row & (S_LEN - 1);
  const int tid = threadIdx.x;
  const float* sp = src + ((size_t)s * BATCH + b) * EMB;
  const float* pp = proj + (size_t)row * EMB;

  float vals[4];
  float lsum = 0.f;
#pragma unroll
  for (int it = 0; it < 4; ++it) {
    const int e = tid + it * 256;
    const float v = sp[e] + pp[e];
    vals[it] = v;
    lsum += v;
  }
  red[tid] = lsum;
  __syncthreads();
  for (int off = 128; off > 0; off >>= 1) {
    if (tid < off) red[tid] += red[tid + off];
    __syncthreads();
  }
  const float mu = red[0] * (1.f / EMB);
  __syncthreads();
  float lvar = 0.f;
#pragma unroll
  for (int it = 0; it < 4; ++it) {
    const float d = vals[it] - mu;
    lvar += d * d;
  }
  red[tid] = lvar;
  __syncthreads();
  for (int off = 128; off > 0; off >>= 1) {
    if (tid < off) red[tid] += red[tid + off];
    __syncthreads();
  }
  const float rstd = rsqrtf(red[0] * (1.f / EMB) + 1e-5f);
#pragma unroll
  for (int it = 0; it < 4; ++it) {
    const int e = tid + it * 256;
    const float o = (vals[it] - mu) * rstd * g[e] + bb[e];
    xf[(size_t)row * EMB + e] = o;
    xb[(size_t)row * EMB + e] = (__bf16)o;
  }
}

__global__ __launch_bounds__(256) void ln2_kernel(
    const float* __restrict__ y, const float* __restrict__ g,
    const float* __restrict__ bb, float* __restrict__ out) {
  __shared__ float red[256];
  const int row = blockIdx.x;       // b*S + s
  const int b = row >> 11;
  const int s = row & (S_LEN - 1);
  const int tid = threadIdx.x;
  const float* yp = y + (size_t)row * EMB;

  float vals[4];
  float lsum = 0.f;
#pragma unroll
  for (int it = 0; it < 4; ++it) {
    const float v = yp[tid + it * 256];
    vals[it] = v;
    lsum += v;
  }
  red[tid] = lsum;
  __syncthreads();
  for (int off = 128; off > 0; off >>= 1) {
    if (tid < off) red[tid] += red[tid + off];
    __syncthreads();
  }
  const float mu = red[0] * (1.f / EMB);
  __syncthreads();
  float lvar = 0.f;
#pragma unroll
  for (int it = 0; it < 4; ++it) {
    const float d = vals[it] - mu;
    lvar += d * d;
  }
  red[tid] = lvar;
  __syncthreads();
  for (int off = 128; off > 0; off >>= 1) {
    if (tid < off) red[tid] += red[tid + off];
    __syncthreads();
  }
  const float rstd = rsqrtf(red[0] * (1.f / EMB) + 1e-5f);
#pragma unroll
  for (int it = 0; it < 4; ++it) {
    const int e = tid + it * 256;
    const float o = (vals[it] - mu) * rstd * g[e] + bb[e];
    out[((size_t)s * BATCH + b) * EMB + e] = o;
  }
}

// ---------------------------------------------------------------------------
// Launch
// ---------------------------------------------------------------------------
extern "C" void kernel_launch(void* const* d_in, const int* in_sizes, int n_in,
                              void* d_out, int out_size, void* d_ws, size_t ws_size,
                              hipStream_t stream) {
  const float* src   = (const float*)d_in[0];
  const float* mask  = (const float*)d_in[1];
  const unsigned char* pad = (const unsigned char*)d_in[2];
  const float* Wq    = (const float*)d_in[3];
  const float* Wk    = (const float*)d_in[4];
  const float* Wv    = (const float*)d_in[5];
  const float* projW = (const float*)d_in[6];
  const float* projb = (const float*)d_in[7];
  const float* W1    = (const float*)d_in[8];
  const float* b1    = (const float*)d_in[9];
  const float* W2    = (const float*)d_in[10];
  const float* b2    = (const float*)d_in[11];
  const float* ln1g  = (const float*)d_in[12];
  const float* ln1b  = (const float*)d_in[13];
  const float* ln2g  = (const float*)d_in[14];
  const float* ln2b  = (const float*)d_in[15];
  float* out = (float*)d_out;

  char* ws = (char*)d_ws;
  size_t o = 0;
  auto carve = [&](size_t bytes) {
    void* p = ws + o;
    o += (bytes + 255) & ~(size_t)255;
    return p;
  };
  __bf16* xb_bf  = (__bf16*)carve((size_t)MROWS * EMB * 2);
  __bf16* wq_bf  = (__bf16*)carve((size_t)EMB * EMB * 2);
  __bf16* wk_bf  = (__bf16*)carve((size_t)EMB * EMB * 2);
  __bf16* wv_bf  = (__bf16*)carve((size_t)EMB * EMB * 2);
  __bf16* pw_bf  = (__bf16*)carve((size_t)EMB * EMB * 2);
  __bf16* w1t_bf = (__bf16*)carve((size_t)EMB * DFF * 2);
  __bf16* w2t_bf = (__bf16*)carve((size_t)EMB * DFF * 2);
  __bf16* q_bf   = (__bf16*)carve((size_t)MROWS * EMB * 2);
  __bf16* k_bf   = (__bf16*)carve((size_t)MROWS * EMB * 2);
  __bf16* vT_bf  = (__bf16*)carve((size_t)MROWS * EMB * 2);
  __bf16* at_bf  = (__bf16*)carve((size_t)MROWS * EMB * 2);
  float*  projo  = (float*)carve((size_t)MROWS * EMB * 4);
  float*  xln_f  = (float*)carve((size_t)MROWS * EMB * 4);
  __bf16* xln_b  = (__bf16*)carve((size_t)MROWS * EMB * 2);
  __bf16* h1_bf  = (__bf16*)carve((size_t)MROWS * DFF * 2);
  float*  y_f    = (float*)carve((size_t)MROWS * EMB * 4);

  const int T = 256;
  // conversions
  cvt_src_kernel<<<(MROWS * EMB + T - 1) / T, T, 0, stream>>>(src, xb_bf);
  cvt_qkvw_kernel<<<(EMB * EMB + T - 1) / T, T, 0, stream>>>(Wq, wq_bf);
  cvt_qkvw_kernel<<<(EMB * EMB + T - 1) / T, T, 0, stream>>>(Wk, wk_bf);
  cvt_qkvw_kernel<<<(EMB * EMB + T - 1) / T, T, 0, stream>>>(Wv, wv_bf);
  cvt_copy_kernel<<<(EMB * EMB + T - 1) / T, T, 0, stream>>>(projW, pw_bf, EMB * EMB);
  cvt_T_kernel<<<(EMB * DFF + T - 1) / T, T, 0, stream>>>(W1, w1t_bf, EMB, DFF);
  cvt_T_kernel<<<(EMB * DFF + T - 1) / T, T, 0, stream>>>(W2, w2t_bf, DFF, EMB);

  // QKV projections (M=4096, N=1024, K=1024)
  dim3 blk(256);
  dim3 g_qkv(EMB / 128, MROWS / 128);
  gemm_bf16_kernel<16><<<g_qkv, blk, 0, stream>>>(xb_bf, wq_bf, nullptr, q_bf,
                                                  nullptr, nullptr, MROWS, EMB, EMB);
  gemm_bf16_kernel<16><<<g_qkv, blk, 0, stream>>>(xb_bf, wk_bf, nullptr, k_bf,
                                                  nullptr, nullptr, MROWS, EMB, EMB);
  gemm_bf16_kernel<32><<<g_qkv, blk, 0, stream>>>(xb_bf, wv_bf, nullptr, vT_bf,
                                                  nullptr, nullptr, MROWS, EMB, EMB);

  // flash attention: B*H*(S/64) blocks of 128 threads
  flash_attn_kernel<<<BATCH * HEADS * (S_LEN / 64), 128, 0, stream>>>(
      q_bf, k_bf, vT_bf, mask, pad, at_bf);

  // output projection: bias, fp32 out
  gemm_bf16_kernel<8 | 1><<<g_qkv, blk, 0, stream>>>(at_bf, pw_bf, projo, nullptr,
                                                     projb, nullptr, MROWS, EMB, EMB);

  // residual + LN1
  add_ln1_kernel<<<MROWS, 256, 0, stream>>>(src, projo, ln1g, ln1b, xln_f, xln_b);

  // FFN1: bias + relu, bf16 out  (M=4096, N=4096, K=1024)
  dim3 g_ff1(DFF / 128, MROWS / 128);
  gemm_bf16_kernel<16 | 2 | 1><<<g_ff1, blk, 0, stream>>>(
      xln_b, w1t_bf, nullptr, h1_bf, b1, nullptr, MROWS, DFF, EMB);

  // FFN2: bias + residual(xln), fp32 out  (M=4096, N=1024, K=4096)
  gemm_bf16_kernel<8 | 4 | 1><<<g_qkv, blk, 0, stream>>>(
      h1_bf, w2t_bf, y_f, nullptr, b2, xln_f, MROWS, EMB, DFF);

  // LN2 + transpose to [S,B,E]
  ln2_kernel<<<MROWS, 256, 0, stream>>>(y_f, ln2g, ln2b, out);

  (void)in_sizes; (void)n_in; (void)out_size; (void)ws_size;
}
